// CausalSelfAttention_15350213115929
// MI455X (gfx1250) — matrix operand order repaired and checked by reading, not compiled, not run on hardware
//
#include <hip/hip_runtime.h>

typedef __attribute__((ext_vector_type(16))) _Float16 v16h;
typedef __attribute__((ext_vector_type(8)))  float    v8f;

#define WMMA_F32_F16(a, b, c) \
  __builtin_amdgcn_wmma_f32_16x16x32_f16(false, (a), false, (b), (short)0, (c), false, false)

static __device__ inline v8f vzero8() {
  v8f z = {0.f, 0.f, 0.f, 0.f, 0.f, 0.f, 0.f, 0.f};
  return z;
}

// log2(10000)/64 : theta_pair = 2^(-(d&~1) * L2B64)
#define L2B64 0.2076205060f

// ---------------------------------------------------------------------------
// Kernel 1: qkv = x @ W_attn + b_attn, fused RoPE (angles indexed by head),
// scatter to q/k [B,H,T,hd] (f16, q pre-scaled by 1/sqrt(hd)) and
// v transposed [B,H,hd,T] (f16).
// Wave computes a 16(M) x 64(N) tile; W tile staged fp32->f16 in LDS, [n][k].
// ---------------------------------------------------------------------------
__global__ __launch_bounds__(256) void qkv_rope_kernel(
    const float* __restrict__ x, const float* __restrict__ W,
    const float* __restrict__ bias,
    _Float16* __restrict__ q_ws, _Float16* __restrict__ k_ws,
    _Float16* __restrict__ v_t) {
  __shared__ _Float16 ldsB[64 * 32];  // [n_rel][k_rel]
  const int tid  = threadIdx.x;
  const int wave = tid >> 5;
  const int lane = tid & 31;
  const int half = lane >> 4;
  const int l16  = lane & 15;
  const int nBase = blockIdx.x * 64;
  const int mBase = blockIdx.y * 128 + wave * 16;

  v8f acc[4] = {vzero8(), vzero8(), vzero8(), vzero8()};

  const float* xrow = x + (size_t)(mBase + l16) * 1024;
  const int ldK = tid >> 3;        // 0..31: k row this thread stages
  const int ldN = (tid & 7) * 8;   // 0..56: first of 8 n columns

  for (int k0 = 0; k0 < 1024; k0 += 32) {
    __syncthreads();
    {
      const float* wp = W + (size_t)(k0 + ldK) * 3072 + nBase + ldN;
      if (k0 + 32 < 1024) __builtin_prefetch(wp + 32 * 3072, 0, 3);
#pragma unroll
      for (int j = 0; j < 8; ++j)
        ldsB[(ldN + j) * 32 + ldK] = (_Float16)wp[j];
    }
    __syncthreads();

    v16h a;
    {
      const float* ap = xrow + k0 + half * 16;
#pragma unroll
      for (int j = 0; j < 8; ++j) {
        a[j]     = (_Float16)ap[j];
        a[8 + j] = (_Float16)ap[8 + j];
      }
    }
#pragma unroll
    for (int t = 0; t < 4; ++t) {
      v16h bf;
      const _Float16* bp = &ldsB[(t * 16 + l16) * 32 + half * 16];
#pragma unroll
      for (int j = 0; j < 16; ++j) bf[j] = bp[j];
      acc[t] = WMMA_F32_F16(a, bf, acc[t]);
    }
  }

  // Epilogue: C layout = lane owns column n (= l16 of tile), rows i + 8*half.
  const int rowBase = mBase + half * 8;
#pragma unroll
  for (int t = 0; t < 4; ++t) {
    const int nGlob = nBase + t * 16 + l16;
    const float bv  = bias[nGlob];
    const int third = nGlob >> 10;   // 0=q, 1=k, 2=v (uniform over tile)
    const int c  = nGlob & 1023;
    const int h  = c >> 6;
    const int d  = c & 63;
    // theta = 10000^(-(d&~1)/64) = exp2(-(d&~1)*log2(10000)/64): one v_exp_f32
    const float theta = exp2f(-(float)(d & ~1) * L2B64);
    const float ang = (float)h * theta;
    const float cs = __cosf(ang), sn = __sinf(ang);
    const float postScale = (third == 0) ? 0.125f : 1.0f;  // fold 1/sqrt(64) into q
#pragma unroll
    for (int i = 0; i < 8; ++i) {
      const int m  = rowBase + i;
      const int bb = m >> 11;
      const int tt = m & 2047;
      float v = acc[t][i] + bv;
      if (third == 2) {
        v_t[(((size_t)(bb * 16 + h)) * 64 + d) * 2048 + tt] = (_Float16)v;
      } else {
        const float p = __shfl_xor(v, 1);  // partner dim lives in adjacent lane
        const float r = (d & 1) ? (v * cs + p * sn) : (v * cs - p * sn);
        _Float16* dst = (third == 0) ? q_ws : k_ws;
        dst[(((size_t)(bb * 16 + h)) * 2048 + tt) * 64 + d] = (_Float16)(r * postScale);
      }
    }
  }
}

// ---------------------------------------------------------------------------
// Kernel 2: flash attention, one wave per 16-query tile, 32-key blocks.
// S = Q(16x64) @ K^T via 2x2 WMMA, online softmax, P@V via 4 WMMA.
// P transposed C-layout -> A-layout through a per-wave LDS bounce.
// Q already carries the 1/sqrt(hd) scale.
// ---------------------------------------------------------------------------
__global__ __launch_bounds__(256) void flash_attn_kernel(
    const _Float16* __restrict__ q_ws, const _Float16* __restrict__ k_ws,
    const _Float16* __restrict__ v_t, _Float16* __restrict__ y_ws) {
  __shared__ _Float16 ldsP[8][16 * 32];
  const int tid  = threadIdx.x;
  const int wave = tid >> 5;
  const int lane = tid & 31;
  const int half = lane >> 4;
  const int l16  = lane & 15;
  const int hh = blockIdx.y;
  const int bb = blockIdx.z;
  const int qBase = (blockIdx.x * 8 + wave) * 16;

  const _Float16* qp = q_ws + (size_t)(bb * 16 + hh) * 2048 * 64;
  const _Float16* kp = k_ws + (size_t)(bb * 16 + hh) * 2048 * 64;
  const _Float16* vp = v_t  + (size_t)(bb * 16 + hh) * 64 * 2048;
  _Float16* myP = &ldsP[wave][0];

  v16h qa[2];
#pragma unroll
  for (int f = 0; f < 2; ++f) {
    const _Float16* ap = qp + (size_t)(qBase + l16) * 64 + f * 32 + half * 16;
#pragma unroll
    for (int j = 0; j < 8; ++j) { qa[f][j] = ap[j]; qa[f][8 + j] = ap[8 + j]; }
  }

  v8f o[4] = {vzero8(), vzero8(), vzero8(), vzero8()};
  float mrow[8], lrow[8];
#pragma unroll
  for (int i = 0; i < 8; ++i) { mrow[i] = -1e30f; lrow[i] = 0.f; }

  for (int kb = 0; kb < qBase + 16; kb += 32) {
    v8f S[2];
#pragma unroll
    for (int s = 0; s < 2; ++s) {
      v16h bk0, bk1;
      const _Float16* kpp = kp + (size_t)(kb + s * 16 + l16) * 64;
#pragma unroll
      for (int j = 0; j < 16; ++j) {
        bk0[j] = kpp[half * 16 + j];
        bk1[j] = kpp[32 + half * 16 + j];
      }
      v8f c = vzero8();
      c = WMMA_F32_F16(qa[0], bk0, c);
      c = WMMA_F32_F16(qa[1], bk1, c);
      S[s] = c;
    }

    float mblk[8];
#pragma unroll
    for (int i = 0; i < 8; ++i) {
      const int qrow = qBase + half * 8 + i;
#pragma unroll
      for (int s = 0; s < 2; ++s) {
        const int key = kb + s * 16 + l16;
        float v = S[s][i];
        if (key > qrow) v = -1e30f;           // causal mask
        S[s][i] = v;
      }
      float t = fmaxf(S[0][i], S[1][i]);
      t = fmaxf(t, __shfl_xor(t, 1));
      t = fmaxf(t, __shfl_xor(t, 2));
      t = fmaxf(t, __shfl_xor(t, 4));
      t = fmaxf(t, __shfl_xor(t, 8));
      mblk[i] = t;
    }
#pragma unroll
    for (int i = 0; i < 8; ++i) {
      const float mnew  = fmaxf(mrow[i], mblk[i]);
      const float alpha = __expf(mrow[i] - mnew);
      const float p0 = __expf(S[0][i] - mnew);
      const float p1 = __expf(S[1][i] - mnew);
      S[0][i] = p0; S[1][i] = p1;
      float rs = p0 + p1;
      rs += __shfl_xor(rs, 1);
      rs += __shfl_xor(rs, 2);
      rs += __shfl_xor(rs, 4);
      rs += __shfl_xor(rs, 8);
      lrow[i] = lrow[i] * alpha + rs;
      mrow[i] = mnew;
#pragma unroll
      for (int t = 0; t < 4; ++t) o[t][i] *= alpha;
    }

    // P: C-layout -> LDS -> A-layout fragment (wave-private region, in-order LDS)
#pragma unroll
    for (int s = 0; s < 2; ++s)
#pragma unroll
      for (int i = 0; i < 8; ++i)
        myP[(half * 8 + i) * 32 + s * 16 + l16] = (_Float16)S[s][i];
    asm volatile("s_wait_dscnt 0" ::: "memory");
    v16h pa;
    {
      const _Float16* pr = myP + l16 * 32 + half * 16;
#pragma unroll
      for (int j = 0; j < 8; ++j) { pa[j] = pr[j]; pa[8 + j] = pr[8 + j]; }
    }

#pragma unroll
    for (int t = 0; t < 4; ++t) {
      v16h bv;
      const _Float16* vpp = vp + (size_t)(t * 16 + l16) * 2048 + kb + half * 16;
#pragma unroll
      for (int j = 0; j < 16; ++j) bv[j] = vpp[j];
      o[t] = WMMA_F32_F16(pa, bv, o[t]);
    }
  }

#pragma unroll
  for (int t = 0; t < 4; ++t)
#pragma unroll
    for (int i = 0; i < 8; ++i) {
      const int qrow = qBase + half * 8 + i;
      y_ws[((size_t)(bb * 2048 + qrow)) * 1024 + hh * 64 + t * 16 + l16] =
          (_Float16)(o[t][i] / lrow[i]);
    }
}

// ---------------------------------------------------------------------------
// Kernel 3: out = y @ W_proj + b_proj (f16 A direct, W staged fp32->f16 LDS)
// ---------------------------------------------------------------------------
__global__ __launch_bounds__(256) void proj_kernel(
    const _Float16* __restrict__ y, const float* __restrict__ W,
    const float* __restrict__ bias, float* __restrict__ out) {
  __shared__ _Float16 ldsB[64 * 32];
  const int tid  = threadIdx.x;
  const int wave = tid >> 5;
  const int lane = tid & 31;
  const int half = lane >> 4;
  const int l16  = lane & 15;
  const int nBase = blockIdx.x * 64;
  const int mBase = blockIdx.y * 128 + wave * 16;

  v8f acc[4] = {vzero8(), vzero8(), vzero8(), vzero8()};

  const _Float16* yrow = y + (size_t)(mBase + l16) * 1024;
  const int ldK = tid >> 3;
  const int ldN = (tid & 7) * 8;

  for (int k0 = 0; k0 < 1024; k0 += 32) {
    __syncthreads();
    {
      const float* wp = W + (size_t)(k0 + ldK) * 1024 + nBase + ldN;
      if (k0 + 32 < 1024) __builtin_prefetch(wp + 32 * 1024, 0, 3);
#pragma unroll
      for (int j = 0; j < 8; ++j)
        ldsB[(ldN + j) * 32 + ldK] = (_Float16)wp[j];
    }
    __syncthreads();

    v16h a;
    {
      const _Float16* ap = yrow + k0 + half * 16;
#pragma unroll
      for (int j = 0; j < 8; ++j) { a[j] = ap[j]; a[8 + j] = ap[8 + j]; }
    }
#pragma unroll
    for (int t = 0; t < 4; ++t) {
      v16h bf;
      const _Float16* bp = &ldsB[(t * 16 + l16) * 32 + half * 16];
#pragma unroll
      for (int j = 0; j < 16; ++j) bf[j] = bp[j];
      acc[t] = WMMA_F32_F16(a, bf, acc[t]);
    }
  }

  const int rowBase = mBase + half * 8;
#pragma unroll
  for (int t = 0; t < 4; ++t) {
    const int nGlob = nBase + t * 16 + l16;
    const float bv = bias[nGlob];
#pragma unroll
    for (int i = 0; i < 8; ++i)
      out[(size_t)(rowBase + i) * 1024 + nGlob] = acc[t][i] + bv;
  }
}

extern "C" void kernel_launch(void* const* d_in, const int* in_sizes, int n_in,
                              void* d_out, int out_size, void* d_ws, size_t ws_size,
                              hipStream_t stream) {
  (void)in_sizes; (void)n_in; (void)out_size; (void)ws_size;
  const float* x      = (const float*)d_in[0];
  const float* W_attn = (const float*)d_in[1];
  const float* b_attn = (const float*)d_in[2];
  const float* W_proj = (const float*)d_in[3];
  const float* b_proj = (const float*)d_in[4];
  float* out = (float*)d_out;

  // Workspace: q 8MB | k 8MB | v_t 8MB | y 8MB (f16), all overwritten each call.
  _Float16* q_ws = (_Float16*)d_ws;
  _Float16* k_ws = q_ws + (size_t)4 * 1024 * 1024;
  _Float16* v_t  = k_ws + (size_t)4 * 1024 * 1024;
  _Float16* y_ws = v_t  + (size_t)4 * 1024 * 1024;

  qkv_rope_kernel<<<dim3(48, 32), 256, 0, stream>>>(x, W_attn, b_attn, q_ws, k_ws, v_t);
  flash_attn_kernel<<<dim3(16, 16, 2), 256, 0, stream>>>(q_ws, k_ws, v_t, y_ws);
  proj_kernel<<<dim3(16, 32), 256, 0, stream>>>(y_ws, W_proj, b_proj, out);
}
